// Raindrop_36687610642965
// MI455X (gfx1250) — compile-verified, matrix operand order checked by reading, not run on hardware
//
#include <hip/hip_runtime.h>

typedef __attribute__((ext_vector_type(16))) _Float16 v16h;
typedef __attribute__((ext_vector_type(8)))  float    v8f;

namespace {
constexpr int B_ = 8, T_ = 60, S_ = 36, H_ = 4, E_ = 16, PE_ = 16;
constexpr int AD_ = 12, TA_ = 10, OUT_ = 64;
constexpr int D28 = AD_ + PE_;   // 28
constexpr int EPE = E_ + PE_;    // 32
constexpr int SS  = S_ * S_;     // 1296
constexpr int KEEP_FROM = 648;   // floor(S*S*0.5)
constexpr float EPSf = 1e-5f;

constexpr int N_H   = B_ * H_ * T_ * S_ * E_;    // 1105920
constexpr int N_AL  = B_ * H_ * T_ * SS;         // 2488320
constexpr int N_ADJ = B_ * H_ * SS;              // 41472
constexpr int N_BID = H_ * SS;                   // 5184

// workspace layout (float offsets)
constexpr size_t o_mu   = 0;                 // T
constexpr size_t o_rstd = o_mu   + 64;       // T
constexpr size_t o_cnt  = o_rstd + 64;       // B*S = 288
constexpr size_t o_pe   = o_cnt  + 288;      // B*T*PE = 7680
constexpr size_t o_bid  = o_pe   + 7680;     // 5184
constexpr size_t o_adj  = o_bid  + 5184;     // 41472
constexpr size_t o_beta = o_adj  + 41472;    // 69120
constexpr size_t o_bm   = o_beta + 69120;    // 32
constexpr size_t o_br   = o_bm   + 32;       // 32
constexpr size_t o_o2   = o_br   + 32;       // B*H*S*16 = 18432
constexpr size_t o_hA   = o_o2   + 18432;    // 1105920
constexpr size_t o_hB   = o_hA   + (size_t)N_H;
constexpr size_t o_al   = o_hB   + (size_t)N_H;
// total = o_al + N_AL ~= 4.8M floats (~19 MB)
}

__device__ __forceinline__ float lrelu(float v) { return v > 0.f ? v : 0.01f * v; }

__device__ __forceinline__ void blk_reduce2(float* s1, float* s2, int tid) {
  for (int off = 128; off > 0; off >>= 1) {
    __syncthreads();
    if (tid < off) { s1[tid] += s1[tid + off]; s2[tid] += s2[tid + off]; }
  }
  __syncthreads();
}

// -------- BatchNorm batch stats per timestep channel t --------
__global__ void rd_bnstats(const float* __restrict__ x, float* __restrict__ ws) {
  __shared__ float s1[256], s2[256];
  int t = blockIdx.x, tid = threadIdx.x;
  float a = 0.f, q = 0.f;
  for (int i = tid; i < B_ * S_; i += 256) {
    int b = i / S_, s = i % S_;
    float v = x[(size_t)(b * T_ + t) * S_ + s];
    a += v; q += v * v;
  }
  s1[tid] = a; s2[tid] = q;
  blk_reduce2(s1, s2, tid);
  if (tid == 0) {
    float mu  = s1[0] / (float)(B_ * S_);
    float var = s2[0] / (float)(B_ * S_) - mu * mu;
    ws[o_mu + t]   = mu;
    ws[o_rstd + t] = rsqrtf(var + EPSf);
  }
}

// -------- sinusoidal PE + per-sensor mask counts --------
__global__ void rd_pe_cnt(const float* __restrict__ times, const float* __restrict__ mask,
                          float* __restrict__ ws) {
  int idx = blockIdx.x * 256 + threadIdx.x;
  if (idx < B_ * T_) {
    float tm = times[idx];
    float* p = ws + o_pe + (size_t)idx * PE_;
#pragma unroll
    for (int i = 0; i < PE_ / 2; ++i) {
      float div = expf(-logf(10000.f) * (float)i / (float)(PE_ / 2));
      float ang = tm * div;
      p[2 * i]     = sinf(ang);
      p[2 * i + 1] = cosf(ang);
    }
  }
  if (idx < B_ * S_) {
    int b = idx / S_, s = idx % S_;
    float c = 0.f;
    for (int t = 0; t < T_; ++t) c += mask[(size_t)(b * T_ + t) * S_ + s];
    ws[o_cnt + idx] = c;
  }
}

// -------- observation embedding: h0 = lrelu(xn * W) * mask --------
__global__ void rd_embed(const float* __restrict__ x, const float* __restrict__ mask,
                         const float* __restrict__ bng, const float* __restrict__ bnb,
                         const float* __restrict__ W, float* ws) {
  int idx = blockIdx.x * 256 + threadIdx.x;
  if (idx >= N_H) return;
  int e  = idx % E_;
  int s  = (idx / E_) % S_;
  int t  = (idx / (E_ * S_)) % T_;
  int hh = (idx / (E_ * S_ * T_)) % H_;
  int b  = idx / (E_ * S_ * T_ * H_);
  float xv = x[(size_t)(b * T_ + t) * S_ + s];
  float xn = (xv - ws[o_mu + t]) * ws[o_rstd + t] * bng[t] + bnb[t];
  float v  = lrelu(xn * W[s * (E_ * H_) + hh * E_ + e]) * mask[(size_t)(b * T_ + t) * S_ + s];
  ws[o_hA + idx] = v;
}

// -------- fused per-(b,h,t):  hp = h @ projWᵀ + pb   (WMMA #1, 3x2 tiles)
//                               alpha = lrelu(hp @ catᵀ) (WMMA #2, 3x3 tiles)
// LDS staging; K padded to 32, M/N padded to 48.  9 waves / 288 threads.
__global__ void rd_alpha(const float* hin, const float* __restrict__ projW,
                         const float* __restrict__ pb, const float* __restrict__ attw,
                         const float* pe, float* alpha) {
  __shared__ _Float16 Hs[48 * 32];  // A for GEMM1: h rows (s), cols e (16 valid)
  __shared__ _Float16 Pw[32 * 32];  // Bᵀ for GEMM1: rows n=d (28 valid), cols k=e (16 valid)
  __shared__ _Float16 As[48 * 32];  // A for GEMM2: hp rows (s), cols d (28 valid)
  __shared__ _Float16 Bs[48 * 32];  // Bᵀ for GEMM2: cat rows (j), cols d (28 valid)
  int blk = blockIdx.x;
  int t = blk % T_;
  int h = (blk / T_) % H_;
  int b = blk / (T_ * H_);
  int tid = threadIdx.x;

  const float* hBase  = hin + (size_t)(((b * H_ + h) * T_ + t) * S_) * E_;
  const float* peBase = pe + (size_t)(b * T_ + t) * PE_;
  const float* awBase = attw + (size_t)h * S_ * AD_;

  // gfx1250 prefetch path (global_prefetch_b8): pull the h tile + weights early.
  __builtin_prefetch(hBase, 0, 3);
  __builtin_prefetch(awBase, 0, 3);

  // Phase 1: stage Hs, Pw, Bs (4096 f16 elements).
  for (int idx = tid; idx < 4096; idx += 288) {
    if (idx < 1536) {                       // Hs
      int r = idx >> 5, c = idx & 31;
      Hs[idx] = (r < S_ && c < E_) ? (_Float16)hBase[r * E_ + c] : (_Float16)0.f;
    } else if (idx < 2560) {                // Pw
      int i2 = idx - 1536;
      int n = i2 >> 5, k = i2 & 31;
      Pw[i2] = (n < D28 && k < E_) ? (_Float16)projW[n * E_ + k] : (_Float16)0.f;
    } else {                                // Bs (cat = [attn_w | pe])
      int i3 = idx - 2560;
      int r = i3 >> 5, c = i3 & 31;
      float bv = 0.f;
      if (r < S_ && c < D28)
        bv = (c < AD_) ? awBase[r * AD_ + c] : peBase[c - AD_];
      Bs[i3] = (_Float16)bv;
    }
  }
  __syncthreads();

  int wave = tid >> 5, lane = tid & 31;
  int rl = lane & 15, kb = (lane >> 4) * 8;

  // Phase 2: GEMM1 -> hp tile into As (waves 0..5; uniform wave branch, EXEC full).
  if (wave < 6) {
    int tm = (wave >> 1) * 16, tn = (wave & 1) * 16;
    v16h a, bb;
#pragma unroll
    for (int j = 0; j < 8; ++j) {
      a[j]      = Hs[(tm + rl) * 32 + kb + j];
      a[j + 8]  = Hs[(tm + rl) * 32 + 16 + kb + j];
      bb[j]     = Pw[(tn + rl) * 32 + kb + j];
      bb[j + 8] = Pw[(tn + rl) * 32 + 16 + kb + j];
    }
    v8f c = {0.f, 0.f, 0.f, 0.f, 0.f, 0.f, 0.f, 0.f};
    c = __builtin_amdgcn_wmma_f32_16x16x32_f16(false, a, false, bb, (short)0, c, false, false);
    int mOff = tm + ((lane >> 4) << 3);
    int n = tn + rl;
    float bias = (n < D28) ? pb[n] : 0.f;
#pragma unroll
    for (int r = 0; r < 8; ++r) {
      int m = mOff + r;
      As[m * 32 + n] = (n < D28) ? (_Float16)(c[r] + bias) : (_Float16)0.f;
    }
  }
  __syncthreads();

  // Phase 3: GEMM2 -> alpha tile (all 9 waves, one 16x16 tile each).
  {
    int tm = (wave / 3) * 16, tn = (wave % 3) * 16;
    v16h a, bb;
#pragma unroll
    for (int j = 0; j < 8; ++j) {
      a[j]      = As[(tm + rl) * 32 + kb + j];
      a[j + 8]  = As[(tm + rl) * 32 + 16 + kb + j];
      bb[j]     = Bs[(tn + rl) * 32 + kb + j];
      bb[j + 8] = Bs[(tn + rl) * 32 + 16 + kb + j];
    }
    v8f c = {0.f, 0.f, 0.f, 0.f, 0.f, 0.f, 0.f, 0.f};
    c = __builtin_amdgcn_wmma_f32_16x16x32_f16(false, a, false, bb, (short)0, c, false, false);

    float* aBase = alpha + (size_t)((b * H_ + h) * T_ + t) * SS;
    int mOff = tm + ((lane >> 4) << 3);
    int n = tn + rl;
#pragma unroll
    for (int r = 0; r < 8; ++r) {
      int m = mOff + r;
      if (m < S_ && n < S_) aBase[m * S_ + n] = lrelu(c[r]);
    }
  }
}

// -------- adj = lrelu(sum_t alpha / cnt)  (layer 1; prior adj == 1) --------
__global__ void rd_adj(float* ws) {
  int idx = blockIdx.x * 256 + threadIdx.x;
  if (idx >= N_ADJ) return;
  int j = idx % S_;
  int s = (idx / S_) % S_;
  int h = (idx / SS) % H_;
  int b = idx / (SS * H_);
  const float* al = ws + o_al;
  float acc = 0.f;
  for (int t = 0; t < T_; ++t)
    acc += al[(size_t)(((b * H_ + h) * T_ + t) * S_ + s) * S_ + j];
  ws[o_adj + idx] = lrelu(acc / ws[o_cnt + b * S_ + s]);
}

// -------- prune: keep elements whose stable rank >= 648 per (b,h) --------
__global__ void rd_prune(float* ws) {
  __shared__ float v[SS];
  int bh = blockIdx.x, tid = threadIdx.x;
  float* adj = ws + o_adj + (size_t)bh * SS;
  for (int i = tid; i < SS; i += 256) v[i] = adj[i];
  __syncthreads();
  for (int i = tid; i < SS; i += 256) {
    float vi = v[i];
    int rank = 0;
    for (int j = 0; j < SS; ++j) {
      float vj = v[j];
      rank += (vj < vi) ? 1 : ((vj == vi && j < i) ? 1 : 0);
    }
    if (rank < KEEP_FROM) adj[i] = 0.f;
  }
}

// -------- sim = pair_sim(pruned adj); written to d_out[18432] --------
__global__ void rd_sim(float* ws, float* out) {
  __shared__ float s1[256], s2[256];
  int tid = threadIdx.x;
  float sq = 0.f, t2 = 0.f;
  const float* adj = ws + o_adj;
  for (int e = tid; e < N_BID; e += 256) {
    float tot = 0.f;
    for (int b = 0; b < B_; ++b) {
      float v = adj[(size_t)b * N_BID + e];
      sq += v * v;
      tot += v;
    }
    t2 += tot * tot;
  }
  s1[tid] = sq; s2[tid] = t2;
  blk_reduce2(s1, s2, tid);
  if (tid == 0)
    out[B_ * S_ * OUT_] = ((float)B_ * s1[0] - s2[0]) / 49.f / (float)SS;
}

// -------- bidir[h,s,j] = lrelu(bw[h,s]·bw[h,j]) --------
__global__ void rd_bidir(const float* __restrict__ bw, float* ws) {
  int idx = blockIdx.x * 256 + threadIdx.x;
  if (idx >= N_BID) return;
  int j = idx % S_;
  int s = (idx / S_) % S_;
  int h = idx / SS;
  float acc = 0.f;
#pragma unroll
  for (int d = 0; d < AD_; ++d)
    acc += bw[(h * S_ + s) * AD_ + d] * bw[(h * S_ + j) * AD_ + d];
  ws[o_bid + idx] = lrelu(acc);
}

// -------- message passing (lrelu inside the j-sum: NOT a GEMM) --------
__global__ void rd_msg(const float* hin, float* hout, const float* ws, int use_adj) {
  int idx = blockIdx.x * 256 + threadIdx.x;
  if (idx >= N_H) return;
  int e  = idx % E_;
  int s  = (idx / E_) % S_;
  int t  = (idx / (E_ * S_)) % T_;
  int h  = (idx / (E_ * S_ * T_)) % H_;
  int b  = idx / (E_ * S_ * T_ * H_);
  const float* al = ws + o_al  + (size_t)(((b * H_ + h) * T_ + t) * S_ + s) * S_;
  const float* bd = ws + o_bid + (size_t)(h * S_ + s) * S_;
  const float* ad = ws + o_adj + (size_t)((b * H_ + h) * S_ + s) * S_;
  const float* hj = hin + (size_t)((b * H_ + h) * T_ + t) * S_ * E_ + e;
  __builtin_prefetch(al, 0, 1);
  float acc = 0.f;
  for (int j = 0; j < S_; ++j) {
    float w = bd[j] * al[j] * (use_adj ? ad[j] : 1.f);
    w = lrelu(w);
    acc += lrelu(hj[j * E_] * w);
  }
  hout[idx] = lrelu(acc);
}

// -------- beta[b,h,s,t] = Q[t]·(Σ_u Ws[u] K[u]) + bs  (Ws folded in) --------
__global__ void rd_beta(const float* hfin,
                        const float* __restrict__ Wq, const float* __restrict__ bq,
                        const float* __restrict__ Wk, const float* __restrict__ bk,
                        const float* __restrict__ Wsw, const float* __restrict__ bsb,
                        float* ws) {
  __shared__ float Ksh[T_ * TA_];
  int blk = blockIdx.x;
  int s = blk % S_;
  int h = (blk / S_) % H_;
  int b = blk / (S_ * H_);
  int t = threadIdx.x;
  float Q[TA_];
  if (t < T_) {
    float hc[EPE];
    const float* hv = hfin + (size_t)(((b * H_ + h) * T_ + t) * S_ + s) * E_;
    const float* pv = ws + o_pe + (size_t)(b * T_ + t) * PE_;
#pragma unroll
    for (int d = 0; d < E_; ++d) hc[d] = hv[d];
#pragma unroll
    for (int d = 0; d < PE_; ++d) hc[E_ + d] = pv[d];
#pragma unroll
    for (int q = 0; q < TA_; ++q) {
      float aq = bq[q], ak = bk[q];
#pragma unroll
      for (int d = 0; d < EPE; ++d) {
        aq += hc[d] * Wq[q * EPE + d];
        ak += hc[d] * Wk[q * EPE + d];
      }
      Q[q] = aq;
      Ksh[t * TA_ + q] = ak;
    }
  }
  __syncthreads();
  if (t < T_) {
    float Kw[TA_];
#pragma unroll
    for (int q = 0; q < TA_; ++q) Kw[q] = 0.f;
    for (int u = 0; u < T_; ++u) {
      float w = Wsw[u];
#pragma unroll
      for (int q = 0; q < TA_; ++q) Kw[q] += w * Ksh[u * TA_ + q];
    }
    float beta = bsb[0];
#pragma unroll
    for (int q = 0; q < TA_; ++q) beta += Q[q] * Kw[q];
    ws[o_beta + (size_t)blk * T_ + t] = beta;
  }
}

// -------- beta mean/var per (b,h) over (S,T) --------
__global__ void rd_bstats(float* ws) {
  __shared__ float s1[256], s2[256];
  int bh = blockIdx.x, tid = threadIdx.x;
  const float* bp = ws + o_beta + (size_t)bh * S_ * T_;
  float a = 0.f, q = 0.f;
  for (int i = tid; i < S_ * T_; i += 256) {
    float v = bp[i];
    a += v; q += v * v;
  }
  s1[tid] = a; s2[tid] = q;
  blk_reduce2(s1, s2, tid);
  if (tid == 0) {
    float m = s1[0] / (float)(S_ * T_);
    float var = s2[0] / (float)(S_ * T_) - m * m;
    ws[o_bm + bh] = m;
    ws[o_br + bh] = rsqrtf(var + EPSf);
  }
}

// -------- out1 = lrelu(betaN·Hc); out2 = lrelu(out1@Wse.T + bse) --------
__global__ void rd_out(const float* hfin,
                       const float* __restrict__ lntg, const float* __restrict__ lntb,
                       const float* __restrict__ Wse, const float* __restrict__ bse,
                       float* ws) {
  __shared__ float bN[T_];
  __shared__ float o1[EPE];
  int blk = blockIdx.x;
  int s = blk % S_;
  int h = (blk / S_) % H_;
  int b = blk / (S_ * H_);
  int bh = b * H_ + h;
  int tid = threadIdx.x;
  if (tid < T_) {
    float raw = ws[o_beta + (size_t)blk * T_ + tid];
    bN[tid] = (raw - ws[o_bm + bh]) * ws[o_br + bh] * lntg[s * T_ + tid] + lntb[s * T_ + tid];
  }
  __syncthreads();
  if (tid < EPE) {
    int d = tid;
    float acc = 0.f;
    for (int t = 0; t < T_; ++t) {
      float hcv = (d < E_)
          ? hfin[(size_t)(((b * H_ + h) * T_ + t) * S_ + s) * E_ + d]
          : ws[o_pe + (size_t)(b * T_ + t) * PE_ + (d - E_)];
      acc += bN[t] * hcv;
    }
    o1[d] = lrelu(acc);
  }
  __syncthreads();
  if (tid < OUT_ / H_) {
    float acc = bse[tid];
#pragma unroll
    for (int d = 0; d < EPE; ++d) acc += Wse[tid * EPE + d] * o1[d];
    ws[o_o2 + (size_t)blk * (OUT_ / H_) + tid] = lrelu(acc);
  }
}

// -------- final LN over (S,16) per (b,h) + transpose to (B,S,OUT) --------
__global__ void rd_final(const float* __restrict__ lnsg, const float* __restrict__ lnsb,
                         float* ws, float* out) {
  __shared__ float s1[256], s2[256];
  int bh = blockIdx.x, tid = threadIdx.x;
  const int N = S_ * (OUT_ / H_);  // 576
  const float* o2 = ws + o_o2 + (size_t)bh * N;
  float a = 0.f, q = 0.f;
  for (int i = tid; i < N; i += 256) {
    float v = o2[i];
    a += v; q += v * v;
  }
  s1[tid] = a; s2[tid] = q;
  blk_reduce2(s1, s2, tid);
  float m2  = s1[0] / (float)N;
  float var = s2[0] / (float)N - m2 * m2;
  float r2  = rsqrtf(var + EPSf);
  int b = bh / H_, hh = bh % H_;
  for (int i = tid; i < N; i += 256) {
    int s = i / (OUT_ / H_), d = i % (OUT_ / H_);
    float v = (o2[i] - m2) * r2 * lnsg[s * (OUT_ / H_) + d] + lnsb[s * (OUT_ / H_) + d];
    out[(size_t)(b * S_ + s) * OUT_ + hh * (OUT_ / H_) + d] = v;
  }
}

extern "C" void kernel_launch(void* const* d_in, const int* in_sizes, int n_in,
                              void* d_out, int out_size, void* d_ws, size_t ws_size,
                              hipStream_t stream) {
  (void)in_sizes; (void)n_in; (void)out_size; (void)ws_size;
  const float* x     = (const float*)d_in[0];
  const float* times = (const float*)d_in[1];
  const float* mask  = (const float*)d_in[2];
  const float* bng   = (const float*)d_in[3];
  const float* bnb   = (const float*)d_in[4];
  const float* obsW  = (const float*)d_in[5];
  const float* attnw = (const float*)d_in[6];  // (L,H,S,AD)
  const float* bidw  = (const float*)d_in[7];  // (L,H,S,AD)
  const float* projW = (const float*)d_in[8];
  const float* projb = (const float*)d_in[9];
  const float* Wq    = (const float*)d_in[10];
  const float* bq    = (const float*)d_in[11];
  const float* Wk    = (const float*)d_in[12];
  const float* bk    = (const float*)d_in[13];
  const float* Wsw   = (const float*)d_in[14];
  const float* bsb   = (const float*)d_in[15];
  const float* lntg  = (const float*)d_in[16];
  const float* lntb  = (const float*)d_in[17];
  const float* Wse   = (const float*)d_in[18];
  const float* bse   = (const float*)d_in[19];
  const float* lnsg  = (const float*)d_in[20];
  const float* lnsb  = (const float*)d_in[21];
  float* out = (float*)d_out;
  float* ws  = (float*)d_ws;

  rd_bnstats<<<T_, 256, 0, stream>>>(x, ws);
  rd_pe_cnt<<<2, 256, 0, stream>>>(times, mask, ws);
  rd_embed<<<N_H / 256, 256, 0, stream>>>(x, mask, bng, bnb, obsW, ws);

  size_t hOff[2] = {o_hA, o_hB};
  for (int lay = 0; lay < 2; ++lay) {
    float* hin  = ws + hOff[lay & 1];
    float* hout = ws + hOff[(lay + 1) & 1];
    rd_alpha<<<B_ * H_ * T_, 288, 0, stream>>>(hin, projW, projb,
                                               attnw + (size_t)lay * H_ * S_ * AD_,
                                               ws + o_pe, ws + o_al);
    if (lay) {
      rd_adj<<<N_ADJ / 256, 256, 0, stream>>>(ws);
      rd_prune<<<B_ * H_, 256, 0, stream>>>(ws);
      rd_sim<<<1, 256, 0, stream>>>(ws, out);
    }
    rd_bidir<<<(N_BID + 255) / 256, 256, 0, stream>>>(bidw + (size_t)lay * H_ * S_ * AD_, ws);
    rd_msg<<<N_H / 256, 256, 0, stream>>>(hin, hout, ws, lay);
  }

  float* hfin = ws + o_hA;  // after two swaps final h is back in hA
  rd_beta<<<B_ * H_ * S_, 64, 0, stream>>>(hfin, Wq, bq, Wk, bk, Wsw, bsb, ws);
  rd_bstats<<<B_ * H_, 256, 0, stream>>>(ws);
  rd_out<<<B_ * H_ * S_, 64, 0, stream>>>(hfin, lntg, lntb, Wse, bse, ws);
  rd_final<<<B_ * H_, 256, 0, stream>>>(lnsg, lnsb, ws, out);
}
